// GNNQAModel_47760036331950
// MI455X (gfx1250) — matrix operand order repaired and checked by reading, not compiled, not run on hardware
//
#include <hip/hip_runtime.h>
#include <hip/hip_bf16.h>
#include <cstdint>
#include <cstddef>

// ---- problem constants (match reference) ----
#define F_IN   256
#define H1     4
#define C1     32
#define HC1    128     // H1*C1
#define C2     128
#define QDIM   768
#define NEG_SLOPE 0.2f

// ---- WMMA types (CDNA5 / gfx1250, wave32) ----
typedef __attribute__((ext_vector_type(16))) __bf16 v16bf;
typedef __attribute__((ext_vector_type(8)))  float  v8f;

union ABfrag { uint4 q[2]; unsigned u[8]; v16bf v; };

__device__ __forceinline__ unsigned short f2bf(float f) {
  unsigned u = __float_as_uint(f);
  unsigned r = u + 0x7FFFu + ((u >> 16) & 1u);          // round-to-nearest-even
  if ((u & 0x7F800000u) == 0x7F800000u) r = u;          // inf/nan passthrough
  return (unsigned short)(r >> 16);
}

__device__ __forceinline__ float eluf(float x) { return x > 0.f ? x : (__expf(x) - 1.f); }

__device__ __forceinline__ void atomicMaxF(float* addr, float v) {
  // monotonic int/uint trick; correct for mixed signs, init with -inf bits
  if (v >= 0.f) atomicMax((int*)addr, __float_as_int(v));
  else          atomicMin((unsigned*)addr, __float_as_uint(v));
}

// ------------------------------------------------------------------
// elementwise conversions / fills
// ------------------------------------------------------------------
__global__ void k_f32_to_bf16(const float* __restrict__ in, unsigned short* __restrict__ out, size_t n) {
  size_t i = (size_t)blockIdx.x * blockDim.x + threadIdx.x;
  size_t stride = (size_t)gridDim.x * blockDim.x;
  for (; i < n; i += stride) out[i] = f2bf(in[i]);
}

// out[c*R + r] = bf16(in[r*C + c])  (store W transposed: [Cout, K])
__global__ void k_transpose_bf16(const float* __restrict__ in, unsigned short* __restrict__ out,
                                 int R, int C) {
  long long t = (long long)blockIdx.x * blockDim.x + threadIdx.x;
  if (t >= (long long)R * C) return;
  int r = (int)(t / C), c = (int)(t % C);
  out[(size_t)c * R + r] = f2bf(in[t]);
}

__global__ void k_fill_u32(unsigned* __restrict__ p, unsigned v, size_t n) {
  size_t i = (size_t)blockIdx.x * blockDim.x + threadIdx.x;
  size_t stride = (size_t)gridDim.x * blockDim.x;
  for (; i < n; i += stride) p[i] = v;
}

// ------------------------------------------------------------------
// WMMA GEMM: C[M,Ncol] = A[M,K](bf16,row-major) * Bt[Ncol,K](bf16)^T
// blockDim 256 = 8 waves; wave w owns output columns [16w,16w+16).
// Each wave computes a 64x16 slab (4 row tiles), reusing its B fragment
// across the 4 WMMAs of every k-step. All fragment loads are b128.
//
// bf16 A 16x32 lane layout: lane(hi=lane>>4, r=lane&15) row=r,
//   u[0..3] = K[k0+8*hi .. +7] (16B contig), u[4..7] = same +16 elems.
// bf16 B 32x16 lane layout: col=r, u[0..3] = K[k0+8*hi .. +7], u[4..7] = same +16 elems.
// ------------------------------------------------------------------
__global__ void k_gemm_wmma(const unsigned short* __restrict__ A,
                            const unsigned short* __restrict__ Bt,
                            float* __restrict__ C,
                            int M, int K, int Ncol) {
  int lane = threadIdx.x & 31;
  int wave = threadIdx.x >> 5;
  int colBase = wave * 16;
  if (colBase >= Ncol) return;
  int r  = lane & 15;
  int hi = lane >> 4;
  int rowBase = blockIdx.x * 64;

  // clamp rows for loads (no EXEC-divergent loads); stores are guarded
  size_t arow[4];
#pragma unroll
  for (int t = 0; t < 4; ++t) {
    int rr = rowBase + t * 16 + r;
    if (rr >= M) rr = M - 1;
    arow[t] = (size_t)rr * K;
  }
  size_t brow = (size_t)(colBase + r) * K;

  v8f acc0 = {}, acc1 = {}, acc2 = {}, acc3 = {};
  for (int k0 = 0; k0 < K; k0 += 32) {
    ABfrag b;
    {
      const uint4* bp = (const uint4*)(Bt + brow + k0 + (hi << 3));
      b.q[0] = bp[0];
      b.q[1] = bp[2];
    }
    ABfrag a0, a1, a2, a3;
    {
      const unsigned short* ap = A + arow[0] + k0 + (hi << 3);
      a0.q[0] = *(const uint4*)ap; a0.q[1] = *(const uint4*)(ap + 16);
    }
    {
      const unsigned short* ap = A + arow[1] + k0 + (hi << 3);
      a1.q[0] = *(const uint4*)ap; a1.q[1] = *(const uint4*)(ap + 16);
    }
    {
      const unsigned short* ap = A + arow[2] + k0 + (hi << 3);
      a2.q[0] = *(const uint4*)ap; a2.q[1] = *(const uint4*)(ap + 16);
    }
    {
      const unsigned short* ap = A + arow[3] + k0 + (hi << 3);
      a3.q[0] = *(const uint4*)ap; a3.q[1] = *(const uint4*)(ap + 16);
    }
    acc0 = __builtin_amdgcn_wmma_f32_16x16x32_bf16(false, a0.v, false, b.v, (short)0, acc0, false, false);
    acc1 = __builtin_amdgcn_wmma_f32_16x16x32_bf16(false, a1.v, false, b.v, (short)0, acc1, false, false);
    acc2 = __builtin_amdgcn_wmma_f32_16x16x32_bf16(false, a2.v, false, b.v, (short)0, acc2, false, false);
    acc3 = __builtin_amdgcn_wmma_f32_16x16x32_bf16(false, a3.v, false, b.v, (short)0, acc3, false, false);
  }

  // D layout: lane -> col = colBase + r; VGPR v -> row v + hi*8 within tile
#pragma unroll
  for (int v = 0; v < 8; ++v) {
    int m0 = rowBase + v + hi * 8;
    if (m0 < M)      C[(size_t)m0 * Ncol + colBase + r] = acc0[v];
    int m1 = m0 + 16;
    if (m1 < M)      C[(size_t)m1 * Ncol + colBase + r] = acc1[v];
    int m2 = m0 + 32;
    if (m2 < M)      C[(size_t)m2 * Ncol + colBase + r] = acc2[v];
    int m3 = m0 + 48;
    if (m3 < M)      C[(size_t)m3 * Ncol + colBase + r] = acc3[v];
  }
}

// ------------------------------------------------------------------
// attention coefficients: as/ad[i*heads+h] = sum_c xw[i, h*ch+c] * a[h,c]
// ------------------------------------------------------------------
__global__ void k_attn_alpha(const float* __restrict__ xw,
                             const float* __restrict__ a_src,
                             const float* __restrict__ a_dst,
                             float* __restrict__ as_, float* __restrict__ ad_,
                             int Nn, int heads, int ch) {
  long long t = (long long)blockIdx.x * blockDim.x + threadIdx.x;
  if (t >= (long long)Nn * heads) return;
  int h = (int)(t % heads);
  long long i = t / heads;
  const float* base = xw + (size_t)i * heads * ch + (size_t)h * ch;
  const float* ws = a_src + (size_t)h * ch;
  const float* wd = a_dst + (size_t)h * ch;
  float s = 0.f, d = 0.f;
  for (int c = 0; c < ch; ++c) { float xv = base[c]; s += xv * ws[c]; d += xv * wd[c]; }
  as_[t] = s; ad_[t] = d;
}

// ------------------------------------------------------------------
// edge passes (self loops = edges Ed..Ed+Nn-1)
// ------------------------------------------------------------------
__device__ __forceinline__ float edge_e(const int* src, const int* dst, long long e, int Ed,
                                        int heads, int h,
                                        const float* as_, const float* ad_, int* dOut) {
  int s = (e < Ed) ? src[e] : (int)(e - Ed);
  int d = (e < Ed) ? dst[e] : (int)(e - Ed);
  *dOut = d;
  float v = as_[(size_t)s * heads + h] + ad_[(size_t)d * heads + h];
  return v > 0.f ? v : NEG_SLOPE * v;
}

__global__ void k_edge_max(const int* __restrict__ src, const int* __restrict__ dst,
                           int Ed, int Nn, int heads,
                           const float* __restrict__ as_, const float* __restrict__ ad_,
                           float* __restrict__ mx) {
  long long t = (long long)blockIdx.x * blockDim.x + threadIdx.x;
  if (t >= (long long)(Ed + Nn) * heads) return;
  int h = (int)(t % heads);
  long long e = t / heads;
  int d; float v = edge_e(src, dst, e, Ed, heads, h, as_, ad_, &d);
  atomicMaxF(&mx[(size_t)d * heads + h], v);
}

__global__ void k_edge_den(const int* __restrict__ src, const int* __restrict__ dst,
                           int Ed, int Nn, int heads,
                           const float* __restrict__ as_, const float* __restrict__ ad_,
                           const float* __restrict__ mx, float* __restrict__ dn) {
  long long t = (long long)blockIdx.x * blockDim.x + threadIdx.x;
  if (t >= (long long)(Ed + Nn) * heads) return;
  int h = (int)(t % heads);
  long long e = t / heads;
  int d; float v = edge_e(src, dst, e, Ed, heads, h, as_, ad_, &d);
  atomicAdd(&dn[(size_t)d * heads + h], __expf(v - mx[(size_t)d * heads + h]));
}

__global__ void k_edge_msg(const int* __restrict__ src, const int* __restrict__ dst,
                           int Ed, int Nn, int heads, int ch,
                           const float* __restrict__ as_, const float* __restrict__ ad_,
                           const float* __restrict__ mx, const float* __restrict__ dn,
                           const float* __restrict__ xw, float* __restrict__ acc) {
  int chunks = ch >> 2;
  long long t = (long long)blockIdx.x * blockDim.x + threadIdx.x;
  long long total = (long long)(Ed + Nn) * heads * chunks;
  if (t >= total) return;
  int q = (int)(t % chunks);
  long long r = t / chunks;
  int h = (int)(r % heads);
  long long e = r / heads;
  int s = (e < Ed) ? src[e] : (int)(e - Ed);
  int d = (e < Ed) ? dst[e] : (int)(e - Ed);
  float v = as_[(size_t)s * heads + h] + ad_[(size_t)d * heads + h];
  v = v > 0.f ? v : NEG_SLOPE * v;
  float alpha = __expf(v - mx[(size_t)d * heads + h]) / (dn[(size_t)d * heads + h] + 1e-16f);
  const float4 xv = *(const float4*)(xw + ((size_t)s * heads + h) * ch + (q << 2));
  float* ap = acc + ((size_t)d * heads + h) * ch + (q << 2);
  atomicAdd(ap + 0, alpha * xv.x);
  atomicAdd(ap + 1, alpha * xv.y);
  atomicAdd(ap + 2, alpha * xv.z);
  atomicAdd(ap + 3, alpha * xv.w);
}

// ------------------------------------------------------------------
// bias + ELU, two output flavors
// ------------------------------------------------------------------
__global__ void k_bias_elu_bf16(const float* __restrict__ acc, const float* __restrict__ bias,
                                unsigned short* __restrict__ out, int Nn, int F) {
  long long t = (long long)blockIdx.x * blockDim.x + threadIdx.x;
  if (t >= (long long)Nn * F) return;
  int c = (int)(t % F);
  out[t] = f2bf(eluf(acc[t] + bias[c]));
}

__global__ void k_bias_elu_f32(const float* __restrict__ acc, const float* __restrict__ bias,
                               float* __restrict__ out, int Nn, int F) {
  long long t = (long long)blockIdx.x * blockDim.x + threadIdx.x;
  if (t >= (long long)Nn * F) return;
  int c = (int)(t % F);
  out[t] = eluf(acc[t] + bias[c]);
}

// ------------------------------------------------------------------
// readout
// ------------------------------------------------------------------
__global__ void k_qdot(const float* __restrict__ q, const float* __restrict__ Wq,
                       const float* __restrict__ bq, float* __restrict__ out) {
  __shared__ float red[256];
  float s = 0.f;
  for (int j = threadIdx.x; j < QDIM; j += 256) s += q[j] * Wq[C2 + j];
  red[threadIdx.x] = s; __syncthreads();
  for (int w = 128; w > 0; w >>= 1) {
    if ((int)threadIdx.x < w) red[threadIdx.x] += red[threadIdx.x + w];
    __syncthreads();
  }
  if (threadIdx.x == 0) out[0] = red[0] + bq[0];
}

__global__ void k_logits(const float* __restrict__ ents, const float* __restrict__ Wq,
                         const float* __restrict__ qdot, const int* __restrict__ qmask,
                         float* __restrict__ logits, float* __restrict__ bred, int Nn) {
  __shared__ float red[256];
  int i = blockIdx.x * 256 + threadIdx.x;
  float l = -3.0e38f;
  if (i < Nn) {
    const float* e = ents + (size_t)i * C2;
    float s = 0.f;
#pragma unroll 8
    for (int c = 0; c < C2; ++c) s += e[c] * Wq[c];
    l = s + qdot[0] + ((float)qmask[i] - 1.f) * 1e9f;
    logits[i] = l;
  }
  red[threadIdx.x] = l; __syncthreads();
  for (int w = 128; w > 0; w >>= 1) {
    if ((int)threadIdx.x < w) red[threadIdx.x] = fmaxf(red[threadIdx.x], red[threadIdx.x + w]);
    __syncthreads();
  }
  if (threadIdx.x == 0) bred[blockIdx.x] = red[0];
}

__global__ void k_reduce(const float* __restrict__ in, int n, float* __restrict__ out, int mode) {
  __shared__ float red[256];
  float v = mode ? 0.f : -3.0e38f;
  for (int j = threadIdx.x; j < n; j += 256) v = mode ? (v + in[j]) : fmaxf(v, in[j]);
  red[threadIdx.x] = v; __syncthreads();
  for (int w = 128; w > 0; w >>= 1) {
    if ((int)threadIdx.x < w)
      red[threadIdx.x] = mode ? (red[threadIdx.x] + red[threadIdx.x + w])
                              : fmaxf(red[threadIdx.x], red[threadIdx.x + w]);
    __syncthreads();
  }
  if (threadIdx.x == 0) out[0] = red[0];
}

__global__ void k_sumexp(float* __restrict__ logits, const float* __restrict__ gmax,
                         float* __restrict__ bred, int Nn) {
  __shared__ float red[256];
  int i = blockIdx.x * 256 + threadIdx.x;
  float w = 0.f;
  if (i < Nn) { w = __expf(logits[i] - gmax[0]); logits[i] = w; }
  red[threadIdx.x] = w; __syncthreads();
  for (int s = 128; s > 0; s >>= 1) {
    if ((int)threadIdx.x < s) red[threadIdx.x] += red[threadIdx.x + s];
    __syncthreads();
  }
  if (threadIdx.x == 0) bred[blockIdx.x] = red[0];
}

// blockDim = 128; each block: 128 nodes -> attn, and partial pooled per channel
__global__ void k_attn_pooled(const float* __restrict__ wexp, const float* __restrict__ gsum,
                              const float* __restrict__ ents, float* __restrict__ attn,
                              float* __restrict__ bpool, int Nn) {
  __shared__ float ash[128];
  int base = blockIdx.x * 128;
  int i = base + threadIdx.x;
  float a = 0.f;
  if (i < Nn) { a = wexp[i] / gsum[0]; attn[i] = a; }
  ash[threadIdx.x] = a;
  __syncthreads();
  int c = threadIdx.x;
  int cnt = Nn - base; if (cnt > 128) cnt = 128;
  float s = 0.f;
  for (int j = 0; j < cnt; ++j) s += ash[j] * ents[(size_t)(base + j) * C2 + c];
  bpool[(size_t)blockIdx.x * C2 + c] = s;
}

__global__ void k_pooled_reduce(const float* __restrict__ bpool, int nb, float* __restrict__ pooled) {
  int c = threadIdx.x;
  float s = 0.f;
  for (int b = 0; b < nb; ++b) s += bpool[(size_t)b * C2 + c];
  pooled[c] = s;
}

// ------------------------------------------------------------------
// host
// ------------------------------------------------------------------
extern "C" void kernel_launch(void* const* d_in, const int* in_sizes, int n_in,
                              void* d_out, int out_size, void* d_ws, size_t ws_size,
                              hipStream_t stream) {
  (void)n_in; (void)out_size; (void)ws_size;
  const float* x     = (const float*)d_in[0];
  const int*   eidx  = (const int*)d_in[1];
  const float* qemb  = (const float*)d_in[2];
  const int*   qmask = (const int*)d_in[3];
  // d_in[4] place_entity_mask unused by reference
  const float* W1    = (const float*)d_in[5];
  const float* asrc1 = (const float*)d_in[6];
  const float* adst1 = (const float*)d_in[7];
  const float* b1    = (const float*)d_in[8];
  const float* W2    = (const float*)d_in[9];
  const float* asrc2 = (const float*)d_in[10];
  const float* adst2 = (const float*)d_in[11];
  const float* b2    = (const float*)d_in[12];
  const float* Wq    = (const float*)d_in[13];
  const float* bq    = (const float*)d_in[14];

  const int Nn = in_sizes[3];          // N (mask length)
  const int Ed = in_sizes[1] / 2;      // E
  const int ET = Ed + Nn;              // edges incl. self-loops

  // workspace carve (256B aligned slots)
  char* wsb = (char*)d_ws; size_t off = 0;
  auto carve = [&](size_t bytes) -> char* {
    char* p = wsb + off; off += (bytes + 255) & ~(size_t)255; return p;
  };
  unsigned short* xbf  = (unsigned short*)carve((size_t)Nn * F_IN * 2); // reused as x2 bf16
  unsigned short* w1t  = (unsigned short*)carve((size_t)F_IN * HC1 * 2);
  unsigned short* w2t  = (unsigned short*)carve((size_t)HC1 * C2 * 2);
  float* xw    = (float*)carve((size_t)Nn * HC1 * 4);   // xw1 then xw2
  float* acc   = (float*)carve((size_t)Nn * HC1 * 4);   // acc1 then acc2
  float* as_   = (float*)carve((size_t)Nn * H1 * 4);
  float* ad_   = (float*)carve((size_t)Nn * H1 * 4);
  float* mx    = (float*)carve((size_t)Nn * H1 * 4);
  float* dn    = (float*)carve((size_t)Nn * H1 * 4);
  float* logits= (float*)carve((size_t)Nn * 4);
  float* scal  = (float*)carve(256);                    // [0]=qdot+bq, [1]=gmax, [2]=gsum
  float* bred  = (float*)carve(4096);
  int nb2 = (Nn + 127) / 128;
  float* bpool = (float*)carve((size_t)nb2 * C2 * 4);

  float* pooled = (float*)d_out;
  float* ents   = pooled + C2;
  float* attn   = ents + (size_t)Nn * C2;

  const int TB = 256;
  auto blks = [](long long total, int tb) -> unsigned { return (unsigned)((total + tb - 1) / tb); };

  // --- prep: bf16 operands ---
  k_f32_to_bf16<<<1024, TB, 0, stream>>>(x, xbf, (size_t)Nn * F_IN);
  k_transpose_bf16<<<blks((long long)F_IN * HC1, TB), TB, 0, stream>>>(W1, w1t, F_IN, HC1);
  k_transpose_bf16<<<blks((long long)HC1 * C2, TB), TB, 0, stream>>>(W2, w2t, HC1, C2);

  // --- layer 1 ---
  k_gemm_wmma<<<(Nn + 63) / 64, 256, 0, stream>>>(xbf, w1t, xw, Nn, F_IN, HC1);
  k_attn_alpha<<<blks((long long)Nn * H1, TB), TB, 0, stream>>>(xw, asrc1, adst1, as_, ad_, Nn, H1, C1);
  k_fill_u32<<<512, TB, 0, stream>>>((unsigned*)mx, 0xFF800000u, (size_t)Nn * H1);
  k_fill_u32<<<512, TB, 0, stream>>>((unsigned*)dn, 0u, (size_t)Nn * H1);
  k_fill_u32<<<2048, TB, 0, stream>>>((unsigned*)acc, 0u, (size_t)Nn * HC1);
  k_edge_max<<<blks((long long)ET * H1, TB), TB, 0, stream>>>(eidx, eidx + Ed, Ed, Nn, H1, as_, ad_, mx);
  k_edge_den<<<blks((long long)ET * H1, TB), TB, 0, stream>>>(eidx, eidx + Ed, Ed, Nn, H1, as_, ad_, mx, dn);
  k_edge_msg<<<blks((long long)ET * H1 * (C1 / 4), TB), TB, 0, stream>>>(
      eidx, eidx + Ed, Ed, Nn, H1, C1, as_, ad_, mx, dn, xw, acc);
  k_bias_elu_bf16<<<blks((long long)Nn * HC1, TB), TB, 0, stream>>>(acc, b1, xbf, Nn, HC1);

  // --- layer 2 (heads=1, ch=128) ---
  k_gemm_wmma<<<(Nn + 63) / 64, 256, 0, stream>>>(xbf, w2t, xw, Nn, HC1, C2);
  k_attn_alpha<<<blks((long long)Nn, TB), TB, 0, stream>>>(xw, asrc2, adst2, as_, ad_, Nn, 1, C2);
  k_fill_u32<<<512, TB, 0, stream>>>((unsigned*)mx, 0xFF800000u, (size_t)Nn);
  k_fill_u32<<<512, TB, 0, stream>>>((unsigned*)dn, 0u, (size_t)Nn);
  k_fill_u32<<<2048, TB, 0, stream>>>((unsigned*)acc, 0u, (size_t)Nn * C2);
  k_edge_max<<<blks((long long)ET, TB), TB, 0, stream>>>(eidx, eidx + Ed, Ed, Nn, 1, as_, ad_, mx);
  k_edge_den<<<blks((long long)ET, TB), TB, 0, stream>>>(eidx, eidx + Ed, Ed, Nn, 1, as_, ad_, mx, dn);
  k_edge_msg<<<blks((long long)ET * (C2 / 4), TB), TB, 0, stream>>>(
      eidx, eidx + Ed, Ed, Nn, 1, C2, as_, ad_, mx, dn, xw, acc);
  k_bias_elu_f32<<<blks((long long)Nn * C2, TB), TB, 0, stream>>>(acc, b2, ents, Nn, C2);

  // --- readout ---
  k_qdot<<<1, 256, 0, stream>>>(qemb, Wq, bq, &scal[0]);
  int nb = (Nn + 255) / 256;
  k_logits<<<nb, 256, 0, stream>>>(ents, Wq, &scal[0], qmask, logits, bred, Nn);
  k_reduce<<<1, 256, 0, stream>>>(bred, nb, &scal[1], 0);          // global max
  k_sumexp<<<nb, 256, 0, stream>>>(logits, &scal[1], bred, Nn);
  k_reduce<<<1, 256, 0, stream>>>(bred, nb, &scal[2], 1);          // global sum
  k_attn_pooled<<<nb2, 128, 0, stream>>>(logits, &scal[2], ents, attn, bpool, Nn);
  k_pooled_reduce<<<1, 128, 0, stream>>>(bpool, nb2, pooled);
}
